// DoubleGRU_24713241821345
// MI455X (gfx1250) — matrix operand order repaired and checked by reading, compiled, not run on hardware
//
#include <hip/hip_runtime.h>

// ---------------------------------------------------------------------------
// DoubleGRU fused kernel for gfx1250 (MI455X), round 2.
// B = 131072 rows, S = 128. 13 GEMMs fused, f16 WMMA / f32 accumulate.
// Round-2 change: M=32 rows per wave (each B fragment feeds 2 WMMAs) to halve
// LDS B-operand traffic (the binding resource). 256 rows/block, 3 resident
// weight slots restaged across 5 phases, 3 activation tiles.
// ---------------------------------------------------------------------------

#define LDS_STRIDE 136                 // halfs per 128-wide row (+8 pad)
#define WTILE_H (LDS_STRIDE * 128)     // halfs per weight tile (transposed W)
#define ATILE_H (LDS_STRIDE * 256)     // halfs per 256-row activation tile
#define ROWS 256                       // rows per workgroup (8 waves * 32)

typedef __attribute__((ext_vector_type(16))) _Float16 v16h;
typedef __attribute__((ext_vector_type(8)))  _Float16 v8h;
typedef __attribute__((ext_vector_type(4)))  _Float16 v4h;
typedef __attribute__((ext_vector_type(8)))  float    v8f;
typedef __attribute__((ext_vector_type(4)))  float    v4f;

__device__ __forceinline__ float fast_sigmoid(float x) {
    return 1.0f / (1.0f + __expf(-x));
}

// 256x128 f32 row-major global -> f16 LDS rows (vectorized b128 / b64).
__device__ __forceinline__ void stage_act(_Float16* dst, const float* __restrict__ src) {
    const v4f* s4 = (const v4f*)src;
    for (int i = threadIdx.x; i < ROWS * 128 / 4; i += 256) {
        const int e = i * 4, r = e >> 7, c = e & 127;
        v4f f = s4[i];
        v4h h;
#pragma unroll
        for (int j = 0; j < 4; ++j) h[j] = (_Float16)f[j];
        *(v4h*)(dst + r * LDS_STRIDE + c) = h;   // 8B aligned (c % 4 == 0)
    }
}

// W[k][n] (128x128 f32, row-major) -> transposed f16 LDS: wt[n][k].
__device__ __forceinline__ void stage_weight(_Float16* dst, const float* __restrict__ W) {
    const v4f* s4 = (const v4f*)W;
    for (int i = threadIdx.x; i < 128 * 128 / 4; i += 256) {
        const int e = i * 4, k = e >> 7, n = e & 127;
        v4f f = s4[i];
#pragma unroll
        for (int j = 0; j < 4; ++j) dst[(n + j) * LDS_STRIDE + k] = (_Float16)f[j];
    }
}

// One wave: D[32 rows x 128 cols] (+)= A[32x128] * W[128x128].
// Two 16-row sub-tiles share every B fragment (halves LDS B traffic).
template <bool INIT>
__device__ __forceinline__ void wave_gemm32(v8f acc[2][8], const _Float16* At, int waveRow,
                                            const _Float16* Wt, int lsub, int lhi) {
    // A fragments, ISA 16-bit A layout:
    // lanes 0-15 : row=lane,    elems 0..7 = K 0..7,  8..15 = K 16..23
    // lanes 16-31: row=lane-16, elems 0..7 = K 8..15, 8..15 = K 24..31
    v16h a[4][2];
#pragma unroll
    for (int sub = 0; sub < 2; ++sub) {
        const _Float16* arow = At + (waveRow + sub * 16 + lsub) * LDS_STRIDE + lhi * 8;
#pragma unroll
        for (int ks = 0; ks < 4; ++ks) {
            v8h lo = *(const v8h*)(arow + ks * 32);
            v8h hi = *(const v8h*)(arow + ks * 32 + 16);
#pragma unroll
            for (int e = 0; e < 8; ++e) { a[ks][sub][e] = lo[e]; a[ks][sub][e + 8] = hi[e]; }
        }
    }
#pragma unroll
    for (int nt = 0; nt < 8; ++nt) {
        v8f c0, c1;
        if (INIT) {
#pragma unroll
            for (int e = 0; e < 8; ++e) { c0[e] = 0.0f; c1[e] = 0.0f; }
        } else {
            c0 = acc[0][nt]; c1 = acc[1][nt];
        }
        // B layout: lane n<16 holds K=0..7,16..23 of column n; lane n+16 holds K=8..15,24..31.
        const _Float16* brow = Wt + (lsub + nt * 16) * LDS_STRIDE + lhi * 8;
#pragma unroll
        for (int ks = 0; ks < 4; ++ks) {
            v8h blo = *(const v8h*)(brow + ks * 32);
            v8h bhi = *(const v8h*)(brow + ks * 32 + 16);
            v16h bf;
#pragma unroll
            for (int e = 0; e < 8; ++e) { bf[e] = blo[e]; bf[e + 8] = bhi[e]; }
            c0 = __builtin_amdgcn_wmma_f32_16x16x32_f16(false, a[ks][0], false, bf,
                                                        (short)0, c0, false, false);
            c1 = __builtin_amdgcn_wmma_f32_16x16x32_f16(false, a[ks][1], false, bf,
                                                        (short)0, c1, false, false);
        }
        acc[0][nt] = c0; acc[1][nt] = c1;
    }
}

__global__ __launch_bounds__(256, 1)
void dgru_fused_kernel(const float* __restrict__ x,  const float* __restrict__ old_h,
                       const float* __restrict__ Wx1, const float* __restrict__ Wx2,
                       const float* __restrict__ Wh,  const float* __restrict__ bias,
                       const float* __restrict__ mid, float* __restrict__ out) {
    extern __shared__ char smem_raw[];
    _Float16* wt = (_Float16*)smem_raw;        // 3 weight slots (transposed f16)
    _Float16* T0 = wt + 3 * WTILE_H;           // x        -> mid_h
    _Float16* T1 = T0 + ATILE_H;               // old_h    -> mid_x
    _Float16* T2 = T1 + ATILE_H;               // r*h      -> r2*mid_h
    float*    bsm = (float*)(T2 + ATILE_H);    // 6*128 biases

    const int tid  = threadIdx.x;
    const int lane = tid & 31;
    const int wave = tid >> 5;
    const int lsub = lane & 15;
    const int lhi  = lane >> 4;
    const int waveRow = wave * 32;             // 32 rows per wave
    const long rowBase = (long)blockIdx.x * ROWS;

    // ---- initial staging ----
    for (int i = tid; i < 6 * 128; i += 256) bsm[i] = bias[i];
    stage_act(T0, x     + rowBase * 128);
    stage_act(T1, old_h + rowBase * 128);
    stage_weight(wt + 0 * WTILE_H, Wx1 + 0 * 16384);   // s0 = Wx1[0]
    stage_weight(wt + 1 * WTILE_H, Wh  + 0 * 16384);   // s1 = Wh[0]
    stage_weight(wt + 2 * WTILE_H, Wx1 + 1 * 16384);   // s2 = Wx1[1]
    __syncthreads();

    v8f zacc[2][8], racc[2][8], tacc[2][8];

    // ---- P1: zacc = x@Wx1[0] + h@Wh[0];  racc = x@Wx1[1] ----
    wave_gemm32<true >(zacc, T0, waveRow, wt + 0 * WTILE_H, lsub, lhi);
    wave_gemm32<false>(zacc, T1, waveRow, wt + 1 * WTILE_H, lsub, lhi);
    wave_gemm32<true >(racc, T0, waveRow, wt + 2 * WTILE_H, lsub, lhi);

    __syncthreads();
    stage_weight(wt + 0 * WTILE_H, Wh  + 1 * 16384);   // s0 = Wh[1]
    stage_weight(wt + 1 * WTILE_H, Wx1 + 2 * 16384);   // s1 = Wx1[2]
    stage_weight(wt + 2 * WTILE_H, Wh  + 2 * 16384);   // s2 = Wh[2]
    __syncthreads();

    // ---- P2 ----
    wave_gemm32<false>(racc, T1, waveRow, wt + 0 * WTILE_H, lsub, lhi); // racc += h@Wh[1]
    // z = sig(zacc+b0) (kept in zacc); r = sig(racc+b1); T2 = r * old_h
#pragma unroll
    for (int sub = 0; sub < 2; ++sub)
#pragma unroll
        for (int nt = 0; nt < 8; ++nt) {
            const int c  = lsub + nt * 16;
            const float b0 = bsm[0 * 128 + c];
            const float b1 = bsm[1 * 128 + c];
#pragma unroll
            for (int v = 0; v < 8; ++v) {
                const int r   = waveRow + sub * 16 + v + 8 * lhi;
                const int idx = r * LDS_STRIDE + c;
                zacc[sub][nt][v] = fast_sigmoid(zacc[sub][nt][v] + b0);
                const float rr = fast_sigmoid(racc[sub][nt][v] + b1);
                T2[idx] = (_Float16)(rr * (float)T1[idx]);   // wave-private rows
            }
        }
    wave_gemm32<true >(tacc, T0, waveRow, wt + 1 * WTILE_H, lsub, lhi); // tacc = x@Wx1[2]
    wave_gemm32<false>(tacc, T2, waveRow, wt + 2 * WTILE_H, lsub, lhi); // tacc += (r*h)@Wh[2]
    // h~ = tanh(tacc+b2); mid_h = z*h + (1-z)*h~ -> T0 (x dead)
#pragma unroll
    for (int sub = 0; sub < 2; ++sub)
#pragma unroll
        for (int nt = 0; nt < 8; ++nt) {
            const int c  = lsub + nt * 16;
            const float b2 = bsm[2 * 128 + c];
#pragma unroll
            for (int v = 0; v < 8; ++v) {
                const int r   = waveRow + sub * 16 + v + 8 * lhi;
                const int idx = r * LDS_STRIDE + c;
                const float ht = tanhf(tacc[sub][nt][v] + b2);
                const float z  = zacc[sub][nt][v];
                T0[idx] = (_Float16)(z * (float)T1[idx] + (1.0f - z) * ht);
            }
        }

    __syncthreads();
    stage_weight(wt + 0 * WTILE_H, mid);               // s0 = mid
    stage_weight(wt + 1 * WTILE_H, Wx2 + 0 * 16384);   // s1 = Wx2[0]
    stage_weight(wt + 2 * WTILE_H, Wh  + 0 * 16384);   // s2 = Wh[0] (stage-2 reuse)
    __syncthreads();

    // ---- P3: mid_x = relu(mid_h@mid) -> T1; z2acc = mid_x@Wx2[0] + mid_h@Wh[0] ----
    wave_gemm32<true>(tacc, T0, waveRow, wt + 0 * WTILE_H, lsub, lhi);
#pragma unroll
    for (int sub = 0; sub < 2; ++sub)
#pragma unroll
        for (int nt = 0; nt < 8; ++nt) {
            const int c = lsub + nt * 16;
#pragma unroll
            for (int v = 0; v < 8; ++v) {
                const int r = waveRow + sub * 16 + v + 8 * lhi;
                const float m = tacc[sub][nt][v];
                T1[r * LDS_STRIDE + c] = (_Float16)(m > 0.0f ? m : 0.0f);
            }
        }
    wave_gemm32<true >(zacc, T1, waveRow, wt + 1 * WTILE_H, lsub, lhi);
    wave_gemm32<false>(zacc, T0, waveRow, wt + 2 * WTILE_H, lsub, lhi);

    __syncthreads();
    stage_weight(wt + 0 * WTILE_H, Wx2 + 1 * 16384);   // s0 = Wx2[1]
    stage_weight(wt + 1 * WTILE_H, Wh  + 4 * 16384);   // s1 = Wh[4]
    stage_weight(wt + 2 * WTILE_H, Wx2 + 2 * 16384);   // s2 = Wx2[2]
    __syncthreads();

    // ---- P4: r2acc = mid_x@Wx2[1] + mid_h@Wh[4]; t2 = mid_x@Wx2[2] ----
    wave_gemm32<true >(racc, T1, waveRow, wt + 0 * WTILE_H, lsub, lhi);
    wave_gemm32<false>(racc, T0, waveRow, wt + 1 * WTILE_H, lsub, lhi);
    // z2 = sig(zacc+b0) (kept); r2 = sig(racc+b4); T2 = r2 * mid_h
#pragma unroll
    for (int sub = 0; sub < 2; ++sub)
#pragma unroll
        for (int nt = 0; nt < 8; ++nt) {
            const int c  = lsub + nt * 16;
            const float b0 = bsm[0 * 128 + c];
            const float b4 = bsm[4 * 128 + c];
#pragma unroll
            for (int v = 0; v < 8; ++v) {
                const int r   = waveRow + sub * 16 + v + 8 * lhi;
                const int idx = r * LDS_STRIDE + c;
                zacc[sub][nt][v] = fast_sigmoid(zacc[sub][nt][v] + b0);
                const float r2 = fast_sigmoid(racc[sub][nt][v] + b4);
                T2[idx] = (_Float16)(r2 * (float)T0[idx]);
            }
        }
    wave_gemm32<true>(tacc, T1, waveRow, wt + 2 * WTILE_H, lsub, lhi);

    __syncthreads();
    stage_weight(wt + 0 * WTILE_H, Wh + 5 * 16384);    // s0 = Wh[5]
    __syncthreads();

    // ---- P5: t2 += (r2*mid_h)@Wh[5]; h = z2*mid_h + (1-z2)*tanh(t2+b5) ----
    wave_gemm32<false>(tacc, T2, waveRow, wt + 0 * WTILE_H, lsub, lhi);
#pragma unroll
    for (int sub = 0; sub < 2; ++sub)
#pragma unroll
        for (int nt = 0; nt < 8; ++nt) {
            const int c  = lsub + nt * 16;
            const float b5 = bsm[5 * 128 + c];
#pragma unroll
            for (int v = 0; v < 8; ++v) {
                const int r   = waveRow + sub * 16 + v + 8 * lhi;
                const int idx = r * LDS_STRIDE + c;
                const float h2 = tanhf(tacc[sub][nt][v] + b5);
                const float z2 = zacc[sub][nt][v];
                out[(rowBase + r) * 128 + c] = z2 * (float)T0[idx] + (1.0f - z2) * h2;
            }
        }
}

extern "C" void kernel_launch(void* const* d_in, const int* in_sizes, int n_in,
                              void* d_out, int out_size, void* d_ws, size_t ws_size,
                              hipStream_t stream) {
    const float* x   = (const float*)d_in[0];
    const float* oh  = (const float*)d_in[1];
    const float* Wx1 = (const float*)d_in[2];
    const float* Wx2 = (const float*)d_in[3];
    const float* Wh  = (const float*)d_in[4];
    const float* b   = (const float*)d_in[5];
    const float* mid = (const float*)d_in[6];
    float* out = (float*)d_out;

    // 3 weight tiles + 3 activation tiles (f16) + biases (f32) = 316,416 B
    const size_t shmem = (size_t)(3 * WTILE_H + 3 * ATILE_H) * 2 + 6 * 128 * 4;
    hipFuncSetAttribute((const void*)dgru_fused_kernel,
                        hipFuncAttributeMaxDynamicSharedMemorySize, (int)shmem);

    const int blocks = 131072 / ROWS;  // 512 workgroups, 8 waves, 32 rows/wave
    dgru_fused_kernel<<<blocks, 256, shmem, stream>>>(x, oh, Wx1, Wx2, Wh, b, mid, out);
}
